// DotProductAttention_54073638256762
// MI455X (gfx1250) — matrix operand order repaired and checked by reading, not compiled, not run on hardware
//
#include <hip/hip_runtime.h>

typedef __attribute__((ext_vector_type(16))) _Float16 v16h;
typedef __attribute__((ext_vector_type(8)))  float    v8f;

// Problem constants (match reference)
constexpr int Bn = 4, Hn = 16, Sn = 2048, Dn = 128;

constexpr int QTILE  = 16;              // q rows per WMMA tile
constexpr int TPW    = 2;               // Q tiles per wave (B-operand reuse)
constexpr int WAVES  = 8;               // waves per block (wave32)
constexpr int QBLK   = QTILE * TPW * WAVES;  // 256 q rows per block
constexpr int KT     = 32;              // keys per staged tile

// LDS strides in bytes (16B aligned, padded to spread banks)
constexpr int K_STRIDE = 272;           // per key row: 128 f16 (256B) + pad
constexpr int V_STRIDE = 80;            // per d row of V^T: 32 f16 (64B) + pad
constexpr int P_STRIDE = 80;            // per q row of P: 32 f16 (64B) + pad

constexpr int K_BYTES = KT * K_STRIDE;          // 8704
constexpr int V_BYTES = Dn * V_STRIDE;          // 10240
constexpr int P_BYTES = QTILE * P_STRIDE;       // 1280 per tile
// total LDS: 2*8704 + 2*10240 + 8*2*1280 = 58368 B (of 320 KB/WGP)

union F16x16 {
    v16h     v;
    uint4    q[2];
    _Float16 h[16];
};

// ---- DPP16 cross-lane reduction within each 16-lane half-row ----
// xor1 -> quad_perm [1,0,3,2] = 0xB1 ; xor2 -> quad_perm [2,3,0,1] = 0x4E
// xor4 -> ROW_HALF_MIRROR (0x141) ; xor8 -> ROW_MIRROR (0x140)
template<int CTRL>
__device__ __forceinline__ float dpp_perm(float x) {
    return __builtin_bit_cast(float,
        __builtin_amdgcn_update_dpp(0, __builtin_bit_cast(int, x),
                                    CTRL, 0xf, 0xf, true));
}
__device__ __forceinline__ float row16_max(float x) {
    x = fmaxf(x, dpp_perm<0xB1>(x));
    x = fmaxf(x, dpp_perm<0x4E>(x));
    x = fmaxf(x, dpp_perm<0x141>(x));
    x = fmaxf(x, dpp_perm<0x140>(x));
    return x;
}
__device__ __forceinline__ float row16_sum(float x) {
    x += dpp_perm<0xB1>(x);
    x += dpp_perm<0x4E>(x);
    x += dpp_perm<0x141>(x);
    x += dpp_perm<0x140>(x);
    return x;
}

__global__ __launch_bounds__(256)
void fa_fwd_causal_kernel(const float* __restrict__ Q,
                          const float* __restrict__ K,
                          const float* __restrict__ V,
                          float* __restrict__ O)
{
    // double-buffered K/V staging + per-wave (2-tile) P scratch
    __shared__ __align__(16) unsigned char
        smem[2 * K_BYTES + 2 * V_BYTES + WAVES * TPW * P_BYTES];

    const int tid  = threadIdx.x;
    const int wave = tid >> 5;
    const int lane = tid & 31;
    unsigned char* sP = smem + 2 * K_BYTES + 2 * V_BYTES + wave * (TPW * P_BYTES);

    const int nqb  = Sn / QBLK;              // 8 q-blocks per (b,h)
    const int bh   = blockIdx.x / nqb;
    const int qblk = blockIdx.x % nqb;
    const int q0b  = qblk * QBLK;            // first q row of block
    const int q0   = q0b + wave * (QTILE * TPW);  // first q row of this wave

    const size_t base = (size_t)bh * Sn * Dn;
    const float* Qp = Q + base;
    const float* Kp = K + base;
    const float* Vp = V + base;
    float*       Op = O + base;

    const int half = lane >> 4;   // 0 or 1
    const int l16  = lane & 15;

    // fold 1/sqrt(D_MODEL) and log2(e) into Q so softmax uses exp2
    const float qscale = 0.08838834764831845f * 1.44269504088896340f;

    // ---- Load 2 Q tiles (each 16x128) into f16 A-fragments ----
    // A layout (16x32 f16): lane holds row M=lane&15; K-dim cols:
    //   half 0: {0..7, 16..23}, half 1: {8..15, 24..31} (contiguous 8-packs)
    v16h aq[TPW][4];
    #pragma unroll
    for (int t = 0; t < TPW; ++t) {
        const float* qrow = Qp + (size_t)(q0 + t * QTILE + l16) * Dn;
        const int koff = half * 8;
        #pragma unroll
        for (int c = 0; c < 4; ++c) {
            F16x16 f;
            const float* p0 = qrow + c * 32 + koff;
            const float* p1 = qrow + c * 32 + 16 + koff;
            float4 x0 = *(const float4*)(p0);
            float4 x1 = *(const float4*)(p0 + 4);
            float4 y0 = *(const float4*)(p1);
            float4 y1 = *(const float4*)(p1 + 4);
            f.h[0]  = (_Float16)(x0.x * qscale); f.h[1]  = (_Float16)(x0.y * qscale);
            f.h[2]  = (_Float16)(x0.z * qscale); f.h[3]  = (_Float16)(x0.w * qscale);
            f.h[4]  = (_Float16)(x1.x * qscale); f.h[5]  = (_Float16)(x1.y * qscale);
            f.h[6]  = (_Float16)(x1.z * qscale); f.h[7]  = (_Float16)(x1.w * qscale);
            f.h[8]  = (_Float16)(y0.x * qscale); f.h[9]  = (_Float16)(y0.y * qscale);
            f.h[10] = (_Float16)(y0.z * qscale); f.h[11] = (_Float16)(y0.w * qscale);
            f.h[12] = (_Float16)(y1.x * qscale); f.h[13] = (_Float16)(y1.y * qscale);
            f.h[14] = (_Float16)(y1.z * qscale); f.h[15] = (_Float16)(y1.w * qscale);
            aq[t][c] = f.v;
        }
    }

    // O accumulators: per tile, 8 C/D fragments (16x16 f32 each).
    // C layout: VGPR r, lane n<16 -> (M=r, N=n); lane>=16 -> (M=r+8, N=n-16)
    v8f o_acc[TPW][8];
    float m_i[TPW][8], l_i[TPW][8];
    #pragma unroll
    for (int t = 0; t < TPW; ++t) {
        #pragma unroll
        for (int j = 0; j < 8; ++j) {
            v8f z = {0.f, 0.f, 0.f, 0.f, 0.f, 0.f, 0.f, 0.f};
            o_acc[t][j] = z;
        }
        #pragma unroll
        for (int r = 0; r < 8; ++r) { m_i[t][r] = -__builtin_inff(); l_i[t][r] = 0.f; }
    }

    const int nkt = (q0b + QBLK) / KT;   // causal: block needs keys [0, q0b+256)

    for (int kt = 0; kt < nkt; ++kt) {
        const int k0  = kt * KT;
        const int buf = kt & 1;
        unsigned char* sK = smem + buf * K_BYTES;
        unsigned char* sV = smem + 2 * K_BYTES + buf * V_BYTES;

        // ---- cooperative stage: K row-major f16, V transposed ([d][key]) f16 ----
        // Double buffered: compute(i) precedes stage(i+2) in per-wave program
        // order, so one barrier after staging is sufficient.
        {
            const int kr    = lane;         // key row within tile (0..31)
            const int gk    = k0 + kr;
            const int dbase = wave * 16;    // each wave converts a 16-wide d slab
            const float* krow = Kp + (size_t)gk * Dn + dbase;
            const float* vrow = Vp + (size_t)gk * Dn + dbase;
            F16x16 kk;
            #pragma unroll
            for (int j4 = 0; j4 < 4; ++j4) {
                float4 kx = *(const float4*)(krow + j4 * 4);
                float4 vx = *(const float4*)(vrow + j4 * 4);
                kk.h[j4 * 4 + 0] = (_Float16)kx.x;
                kk.h[j4 * 4 + 1] = (_Float16)kx.y;
                kk.h[j4 * 4 + 2] = (_Float16)kx.z;
                kk.h[j4 * 4 + 3] = (_Float16)kx.w;
                // transpose-at-store: V^T[d][key]
                ((_Float16*)(sV + (dbase + j4 * 4 + 0) * V_STRIDE))[kr] = (_Float16)vx.x;
                ((_Float16*)(sV + (dbase + j4 * 4 + 1) * V_STRIDE))[kr] = (_Float16)vx.y;
                ((_Float16*)(sV + (dbase + j4 * 4 + 2) * V_STRIDE))[kr] = (_Float16)vx.z;
                ((_Float16*)(sV + (dbase + j4 * 4 + 3) * V_STRIDE))[kr] = (_Float16)vx.w;
            }
            *(uint4*)(sK + kr * K_STRIDE + dbase * 2)      = kk.q[0];
            *(uint4*)(sK + kr * K_STRIDE + dbase * 2 + 16) = kk.q[1];

            if (kt + 1 < nkt) {   // prefetch next tile's rows -> global_prefetch
                __builtin_prefetch(krow + KT * Dn, 0, 1);
                __builtin_prefetch(vrow + KT * Dn, 0, 1);
            }
        }
        __syncthreads();

        // Waves past their causal frontier skip compute but keep the barrier.
        // (tile 1 active => compute both; tile 0 fully-masked rows give p=0.)
        if (k0 <= q0 + QTILE * TPW - 1) {
            // ---- S = (Q*scale) K^T : per tile, two 16x16 key halves ----
            // B fragments are loaded once and reused by both Q tiles.
            v8f s0[TPW], s1[TPW];
            #pragma unroll
            for (int t = 0; t < TPW; ++t) {
                v8f z = {0.f, 0.f, 0.f, 0.f, 0.f, 0.f, 0.f, 0.f};
                s0[t] = z; s1[t] = z;
            }
            #pragma unroll
            for (int c = 0; c < 4; ++c) {
                // B layout (32x16 f16): lane = column N (=key), K-dim (=d):
                // half 0: {0..7, 16..23}, half 1: {8..15, 24..31}
                const int doff = (c * 32 + half * 8) * 2;
                F16x16 b0, b1;
                b0.q[0] = *(const uint4*)(sK + l16 * K_STRIDE + doff);
                b0.q[1] = *(const uint4*)(sK + l16 * K_STRIDE + doff + 32);
                b1.q[0] = *(const uint4*)(sK + (l16 + 16) * K_STRIDE + doff);
                b1.q[1] = *(const uint4*)(sK + (l16 + 16) * K_STRIDE + doff + 32);
                #pragma unroll
                for (int t = 0; t < TPW; ++t) {
                    s0[t] = __builtin_amdgcn_wmma_f32_16x16x32_f16(
                            false, aq[t][c], false, b0.v, (short)0, s0[t], false, false);
                    s1[t] = __builtin_amdgcn_wmma_f32_16x16x32_f16(
                            false, aq[t][c], false, b1.v, (short)0, s1[t], false, false);
                }
            }

            // ---- causal mask + online softmax (C layout aware, DPP reductions) ----
            #pragma unroll
            for (int t = 0; t < TPW; ++t) {
                const int qrow_my = q0 + t * QTILE + 8 * half;
                unsigned char* sPt = sP + t * P_BYTES;
                #pragma unroll
                for (int r = 0; r < 8; ++r) {
                    const int qi = qrow_my + r;
                    float a = s0[t][r];
                    float b = s1[t][r];
                    if (k0 + l16      > qi) a = -__builtin_inff();
                    if (k0 + 16 + l16 > qi) b = -__builtin_inff();
                    const float mx   = row16_max(fmaxf(a, b));
                    const float mnew = fmaxf(m_i[t][r], mx);
                    const float p0 = __builtin_amdgcn_exp2f(a - mnew);
                    const float p1 = __builtin_amdgcn_exp2f(b - mnew);
                    const float sum = row16_sum(p0 + p1);
                    const float alpha = __builtin_amdgcn_exp2f(m_i[t][r] - mnew);
                    l_i[t][r] = l_i[t][r] * alpha + sum;
                    m_i[t][r] = mnew;
                    #pragma unroll
                    for (int j = 0; j < 8; ++j) o_acc[t][j][r] *= alpha;
                    // stash P row f16 in this tile's LDS scratch (C->A transpose)
                    _Float16* prow = (_Float16*)(sPt + (r + 8 * half) * P_STRIDE);
                    prow[l16]      = (_Float16)p0;
                    prow[l16 + 16] = (_Float16)p1;
                }
            }

            // ---- reload P as A-fragments (16x32), O += P * V ----
            F16x16 ap[TPW];
            #pragma unroll
            for (int t = 0; t < TPW; ++t) {
                const int koff = half * 8;
                unsigned char* sPt = sP + t * P_BYTES;
                ap[t].q[0] = *(const uint4*)(sPt + l16 * P_STRIDE + koff * 2);
                ap[t].q[1] = *(const uint4*)(sPt + l16 * P_STRIDE + 32 + koff * 2);
            }
            #pragma unroll
            for (int j = 0; j < 8; ++j) {
                // B = V tile: N = d column, K-dim = 32 keys; V^T rows are
                // contiguous in key -> two 16B LDS loads, reused by both tiles
                const unsigned char* vp = sV + (j * 16 + l16) * V_STRIDE + half * 16;
                F16x16 bv;
                bv.q[0] = *(const uint4*)(vp);
                bv.q[1] = *(const uint4*)(vp + 32);
                #pragma unroll
                for (int t = 0; t < TPW; ++t)
                    o_acc[t][j] = __builtin_amdgcn_wmma_f32_16x16x32_f16(
                            false, ap[t].v, false, bv.v, (short)0, o_acc[t][j], false, false);
            }
        }
    }

    // ---- normalize and write out (f32) ----
    #pragma unroll
    for (int t = 0; t < TPW; ++t) {
        #pragma unroll
        for (int r = 0; r < 8; ++r) {
            const float inv = 1.0f / l_i[t][r];
            float* orow = Op + (size_t)(q0 + t * QTILE + r + 8 * half) * Dn;
            #pragma unroll
            for (int j = 0; j < 8; ++j)
                orow[j * 16 + l16] = o_acc[t][j][r] * inv;
        }
    }
}

extern "C" void kernel_launch(void* const* d_in, const int* in_sizes, int n_in,
                              void* d_out, int out_size, void* d_ws, size_t ws_size,
                              hipStream_t stream) {
    (void)in_sizes; (void)n_in; (void)out_size; (void)d_ws; (void)ws_size;
    const float* Q = (const float*)d_in[0];
    const float* K = (const float*)d_in[1];
    const float* V = (const float*)d_in[2];
    float* O = (float*)d_out;

    dim3 grid(Bn * Hn * (Sn / QBLK));   // 512 blocks
    dim3 block(256);                    // 8 wave32 waves per workgroup
    hipLaunchKernelGGL(fa_fwd_causal_kernel, grid, block, 0, stream, Q, K, V, O);
}
